// Mamba2Mixer_21844203668206
// MI455X (gfx1250) — matrix operand order repaired and checked by reading, not compiled, not run on hardware
//
#include <hip/hip_runtime.h>
#include <hip/hip_bf16.h>

// ---------------------------------------------------------------------------
// Mamba2 mixer forward for MI455X (gfx1250, wave32, WMMA).
// All matrix contractions use v_wmma_f32_16x16x32_bf16 (bf16 in, f32 accum).
// ---------------------------------------------------------------------------

typedef __attribute__((ext_vector_type(16))) __bf16 v16bf;
typedef __attribute__((ext_vector_type(8)))  float  v8f;

constexpr int Bb    = 2;
constexpr int Lseq  = 4096;
constexpr int Hh    = 2048;
constexpr int Ii    = 4096;
constexpr int Nn    = 128;
constexpr int HEADS = 64;
constexpr int Pp    = 64;
constexpr int CH    = 256;   // chunk length
constexpr int NC    = 16;    // chunks per sequence
constexpr int ZXW   = 2 * (Ii + Nn) + HEADS;   // 8512
constexpr int XBCW  = Ii + 2 * Nn;             // 4352
constexpr int BL    = Bb * Lseq;               // 8192

__device__ __forceinline__ __bf16 to_bf16(float f) {
  union { float f; unsigned u; } v; v.f = f;
  unsigned r = (v.u + 0x7FFFu + ((v.u >> 16) & 1u)) >> 16;   // RNE
  union { unsigned short s; __bf16 b; } o; o.s = (unsigned short)r;
  return o.b;
}

__device__ __forceinline__ v8f vzero8() {
  v8f v;
#pragma unroll
  for (int i = 0; i < 8; ++i) v[i] = 0.0f;
  return v;
}

__device__ __forceinline__ v8f wmma_bf16(v16bf a, v16bf b, v8f c) {
  // (neg_a, A, neg_b, B, c_mod, C, reuse_a, reuse_b)
  return __builtin_amdgcn_wmma_f32_16x16x32_bf16(false, a, false, b, (short)0, c,
                                                 false, false);
}

// A fragment, 16x32 bf16 tile in LDS, row-major rows=M, ld in elements.
// Layout: lanes 0-15 M=0..15 K=0..15; lanes 16-31 M=0..15 K=16..31 (matches B frags);
// VGPR j packs (k, k+1) in (lo, hi) halves.
__device__ __forceinline__ v16bf frag_a_16x32(const __bf16* tile, int ld, int lane) {
  const int m = lane & 15, half = lane >> 4;
  const __bf16* row = tile + m * ld;
  v16bf f;
#pragma unroll
  for (int j = 0; j < 8; ++j) {
    int kb = 2 * j + 16 * half;
    f[2 * j]     = row[kb];
    f[2 * j + 1] = row[kb + 1];
  }
  return f;
}

// B fragment from a tile stored [n][k] (e.g. weight rows along K), ld in elems.
// Lanes 0-15: N=lane, K=0..15; lanes 16-31: N=lane-16, K=16..31 (k,k+1 packed).
__device__ __forceinline__ v16bf frag_b_nk(const __bf16* tile, int ld, int lane) {
  const int n = lane & 15, koff = (lane >> 4) * 16;
  const __bf16* p = tile + n * ld + koff;
  v16bf f;
#pragma unroll
  for (int j = 0; j < 16; ++j) f[j] = p[j];
  return f;
}

// B fragment from a tile stored [k][n] row-major, ld in elems.
__device__ __forceinline__ v16bf frag_b_kn(const __bf16* tile, int ld, int lane) {
  const int n = lane & 15, koff = (lane >> 4) * 16;
  v16bf f;
#pragma unroll
  for (int j = 0; j < 16; ++j) f[j] = tile[(koff + j) * ld + n];
  return f;
}

// ---------------------------------------------------------------------------
// Generic batched GEMM: C[m,n] = sum_k A[m,k] * W[n,k], f32 in/out, bf16 WMMA.
// 128x128 block tile, 256 threads (8 waves), K step 32, wave tile 32x64.
// ---------------------------------------------------------------------------
__global__ __launch_bounds__(256) void gemm_f32wmma_kernel(
    const float* __restrict__ A, const float* __restrict__ W, float* __restrict__ C,
    int M, int N, int K, long long lda,
    long long batchA, long long batchW, long long batchC) {
  __shared__ __bf16 As[128][40];
  __shared__ __bf16 Ws[128][40];
  const int tid = threadIdx.x;
  const int lane = tid & 31, wave = tid >> 5;
  const int wm = wave & 3, wn = wave >> 2;
  const size_t zb = blockIdx.z;
  A += zb * (size_t)batchA;
  W += zb * (size_t)batchW;
  C += zb * (size_t)batchC;
  const int m0 = blockIdx.y * 128, n0 = blockIdx.x * 128;

  v8f acc[2][4];
#pragma unroll
  for (int i = 0; i < 2; ++i)
#pragma unroll
    for (int j = 0; j < 4; ++j) acc[i][j] = vzero8();

  const int lrow = tid >> 1, lseg = (tid & 1) * 16;
  int am = m0 + lrow; if (am > M - 1) am = M - 1;
  int wr = n0 + lrow; if (wr > N - 1) wr = N - 1;
  const float* Abase = A + (size_t)am * (size_t)lda;
  const float* Wbase = W + (size_t)wr * (size_t)K;

  for (int k0 = 0; k0 < K; k0 += 32) {
    const float4* ap = (const float4*)(Abase + k0 + lseg);
    const float4* wp = (const float4*)(Wbase + k0 + lseg);
#pragma unroll
    for (int i = 0; i < 4; ++i) {
      float4 va = ap[i], vw = wp[i];
      int cc = lseg + i * 4;
      As[lrow][cc + 0] = to_bf16(va.x); As[lrow][cc + 1] = to_bf16(va.y);
      As[lrow][cc + 2] = to_bf16(va.z); As[lrow][cc + 3] = to_bf16(va.w);
      Ws[lrow][cc + 0] = to_bf16(vw.x); Ws[lrow][cc + 1] = to_bf16(vw.y);
      Ws[lrow][cc + 2] = to_bf16(vw.z); Ws[lrow][cc + 3] = to_bf16(vw.w);
    }
    __syncthreads();
    v16bf af[2], bf[4];
    af[0] = frag_a_16x32(&As[wm * 32][0], 40, lane);
    af[1] = frag_a_16x32(&As[wm * 32 + 16][0], 40, lane);
#pragma unroll
    for (int j = 0; j < 4; ++j) bf[j] = frag_b_nk(&Ws[wn * 64 + j * 16][0], 40, lane);
#pragma unroll
    for (int i = 0; i < 2; ++i)
#pragma unroll
      for (int j = 0; j < 4; ++j) acc[i][j] = wmma_bf16(af[i], bf[j], acc[i][j]);
    __syncthreads();
  }

  const int half = lane >> 4, nn = lane & 15;
  if (m0 + 128 <= M && n0 + 128 <= N) {
    // fast path: whole block in bounds, branch-free strided stores
    float* cb = C + (size_t)(m0 + wm * 32 + 8 * half) * (size_t)N + (n0 + wn * 64 + nn);
#pragma unroll
    for (int i = 0; i < 2; ++i)
#pragma unroll
      for (int v = 0; v < 8; ++v) {
        float* cr = cb + (size_t)(i * 16 + v) * (size_t)N;
#pragma unroll
        for (int j = 0; j < 4; ++j) cr[j * 16] = acc[i][j][v];
      }
  } else {
#pragma unroll
    for (int i = 0; i < 2; ++i)
#pragma unroll
      for (int j = 0; j < 4; ++j)
#pragma unroll
        for (int v = 0; v < 8; ++v) {
          int m = m0 + wm * 32 + i * 16 + v + 8 * half;
          int n = n0 + wn * 64 + j * 16 + nn;
          if (m < M && n < N) C[(size_t)m * (size_t)N + n] = acc[i][j][v];
        }
  }
}

// ---------------------------------------------------------------------------
// dt = clip(softplus(dt_raw + dt_bias), 0, 100); dA = dt * (-exp(A_log))
// ---------------------------------------------------------------------------
__global__ __launch_bounds__(256) void dt_kernel(
    const float* __restrict__ zx, const float* __restrict__ dt_bias,
    const float* __restrict__ A_log, float* __restrict__ dt, float* __restrict__ dA) {
  int t = blockIdx.x * 256 + threadIdx.x;           // BL * HEADS = 524288
  int h = t & 63, bl = t >> 6;
  float x = zx[(size_t)bl * ZXW + (Ii + XBCW) + h] + dt_bias[h];
  float sp = (x > 20.0f) ? x : log1pf(__expf(x));
  sp = fminf(fmaxf(sp, 0.0f), 100.0f);
  dt[t] = sp;
  dA[t] = sp * (-__expf(A_log[h]));
}

// ---------------------------------------------------------------------------
// Depthwise causal conv(K=4) + bias + SiLU, split into x / X=x*dt (bf16) / B / C
// ---------------------------------------------------------------------------
__global__ __launch_bounds__(256) void conv_kernel(
    const float* __restrict__ zx, const float* __restrict__ cw,
    const float* __restrict__ cb, const float* __restrict__ dt,
    float* __restrict__ x, __bf16* __restrict__ Xb,
    float* __restrict__ Bf, float* __restrict__ Cf) {
  size_t t = (size_t)blockIdx.x * 256 + threadIdx.x;
  if (t >= (size_t)BL * XBCW) return;
  int ch = (int)(t % XBCW);
  int bl = (int)(t / XBCW);
  int b = bl >> 12, l = bl & 4095;
  float acc = cb[ch];
  const float* w = cw + (size_t)ch * 4;
#pragma unroll
  for (int k = 0; k < 4; ++k) {
    int ls = l - 3 + k;
    if (ls >= 0) acc += w[k] * zx[((size_t)(b * Lseq + ls)) * ZXW + Ii + ch];
  }
  float v = acc / (1.0f + __expf(-acc));            // SiLU
  if (ch < Ii) {
    int hh = ch >> 6;
    x[(size_t)bl * Ii + ch]  = v;
    Xb[(size_t)bl * Ii + ch] = to_bf16(v * dt[(size_t)bl * HEADS + hh]);
  } else if (ch < Ii + Nn) {
    Bf[(size_t)bl * Nn + (ch - Ii)] = v;
  } else {
    Cf[(size_t)bl * Nn + (ch - Ii - Nn)] = v;
  }
}

// ---------------------------------------------------------------------------
// A_cum[b,h,c,:] = cumsum over chunk of dA;  one thread per (b,h,c)
// ---------------------------------------------------------------------------
__global__ __launch_bounds__(256) void cumsum_kernel(
    const float* __restrict__ dA, float* __restrict__ Acum) {
  int t = blockIdx.x * 256 + threadIdx.x;           // 2048
  int c = t & 15, h = (t >> 4) & 63, b = t >> 10;
  float acc = 0.0f;
  const float* src = dA + ((size_t)(b * Lseq + c * CH)) * HEADS + h;
  float* dst = Acum + (((size_t)(b * HEADS + h)) * NC + c) * CH;
  for (int i = 0; i < CH; ++i) { acc += src[(size_t)i * HEADS]; dst[i] = acc; }
}

// ---------------------------------------------------------------------------
// states[b,c,h,p,n] = sum_l exp(Asum - Acum[l]) * X[l,h,p] * B[l,n]
// One block per (h,c,b): 64x128 output, K=256.
// ---------------------------------------------------------------------------
__global__ __launch_bounds__(256) void chunk_states_kernel(
    const __bf16* __restrict__ Xb, const float* __restrict__ Bf,
    const float* __restrict__ Acum, float* __restrict__ states) {
  const int h = blockIdx.x, c = blockIdx.y, b = blockIdx.z;
  __shared__ float sAc[CH];
  __shared__ __bf16 Asm[64][40];
  __shared__ __bf16 Bsm[32][136];
  const int tid = threadIdx.x, lane = tid & 31, wave = tid >> 5;
  const int wm = wave & 3, wn = wave >> 2;
  sAc[tid] = Acum[(((size_t)(b * HEADS + h)) * NC + c) * CH + tid];
  __syncthreads();
  const float Atot = sAc[CH - 1];

  v8f acc[4];
#pragma unroll
  for (int j = 0; j < 4; ++j) acc[j] = vzero8();

  for (int l0 = 0; l0 < CH; l0 += 32) {
    {   // A tile: A[p][kl] = X[l0+kl, h, p] * decay(l0+kl)   (transposed gather)
      int kl = tid >> 3, pg = (tid & 7) * 8;
      float dec = __expf(Atot - sAc[l0 + kl]);
      const __bf16* xp = Xb + ((size_t)(b * Lseq + c * CH + l0 + kl)) * Ii + h * Pp + pg;
#pragma unroll
      for (int j = 0; j < 8; ++j) Asm[pg + j][kl] = to_bf16((float)xp[j] * dec);
    }
    {   // B tile: B[kl][n]
      int kl = tid >> 3, ng = (tid & 7) * 16;
      const float* bp = Bf + ((size_t)(b * Lseq + c * CH + l0 + kl)) * Nn + ng;
#pragma unroll
      for (int j = 0; j < 16; ++j) Bsm[kl][ng + j] = to_bf16(bp[j]);
    }
    __syncthreads();
    v16bf af = frag_a_16x32(&Asm[wm * 16][0], 40, lane);
#pragma unroll
    for (int j = 0; j < 4; ++j) {
      v16bf bf = frag_b_kn(&Bsm[0][wn * 64 + j * 16], 136, lane);
      acc[j] = wmma_bf16(af, bf, acc[j]);
    }
    __syncthreads();
  }

  // branch-free store: always in bounds
  const int half = lane >> 4, nn = lane & 15;
  float* sb = states + (((size_t)((b * NC + c) * HEADS + h)) * Pp) * Nn +
              (size_t)(wm * 16 + 8 * half) * Nn + (wn * 64 + nn);
#pragma unroll
  for (int v = 0; v < 8; ++v) {
    float* sr = sb + (size_t)v * Nn;
#pragma unroll
    for (int j = 0; j < 4; ++j) sr[j * 16] = acc[j][v];
  }
}

// ---------------------------------------------------------------------------
// Inter-chunk recurrence:  prev[0]=0; prev[c+1] = prev[c]*exp(Asum_c) + states[c]
// One thread per (b,h,p,n).
// ---------------------------------------------------------------------------
__global__ __launch_bounds__(256) void prev_states_kernel(
    const float* __restrict__ states, const float* __restrict__ Acum,
    __bf16* __restrict__ prevs) {
  size_t t = (size_t)blockIdx.x * 256 + threadIdx.x;   // 2*64*64*128 = 1048576
  int n = (int)(t & 127);
  int p = (int)((t >> 7) & 63);
  int h = (int)((t >> 13) & 63);
  int b = (int)(t >> 19);
  float run = 0.0f;
  for (int c = 0; c < NC; ++c) {
    size_t idx = ((((size_t)((b * NC + c) * HEADS + h)) * Pp) + p) * Nn + n;
    prevs[idx] = to_bf16(run);
    float Asum = Acum[(((size_t)(b * HEADS + h)) * NC + c) * CH + (CH - 1)];
    run = run * __expf(Asum) + states[idx];
  }
}

// ---------------------------------------------------------------------------
// Fused Y kernel per (h,c,b): 256x64 output
//   phase 1: Y += (G .* L) @ X_h         (K=256, L=exp(Acum[l]-Acum[s]) tri)
//   phase 2: Y += diag(exp(Acum)) C @ prev_hᵀ   (K=128)
//   epilogue: y = Y + x*D[h], written in place over x.
// ---------------------------------------------------------------------------
__global__ __launch_bounds__(256) void chunk_y_kernel(
    const float* __restrict__ G, const __bf16* __restrict__ Xb,
    const float* __restrict__ Cf, const __bf16* __restrict__ prevs,
    const float* __restrict__ Acum, const float* __restrict__ Dvec,
    float* __restrict__ xy) {
  const int h = blockIdx.x, c = blockIdx.y, b = blockIdx.z;
  __shared__ float sAc[CH];
  __shared__ __bf16 Ms[CH][40];     // A-operand K-slab (256 x 32)
  __shared__ __bf16 Ns[32][72];     // B-operand K-slab (32 x 64), [k][n]
  const int tid = threadIdx.x, lane = tid & 31, wave = tid >> 5;
  sAc[tid] = Acum[(((size_t)(b * HEADS + h)) * NC + c) * CH + tid];
  __syncthreads();

  v8f acc[2][4];
#pragma unroll
  for (int i = 0; i < 2; ++i)
#pragma unroll
    for (int j = 0; j < 4; ++j) acc[i][j] = vzero8();

  const size_t gbase = ((size_t)(b * NC + c)) * CH * CH;

  // ---- phase 1: (G .* L) @ X ----
  for (int s0 = 0; s0 < CH; s0 += 32) {
    {   // build M rows: thread tid owns row l = tid
      const float al = sAc[tid];
      const float* gr = G + gbase + (size_t)tid * CH + s0;
#pragma unroll
      for (int i = 0; i < 32; ++i) {
        int s = s0 + i;
        float v = 0.0f;
        if (tid >= s) v = gr[i] * __expf(al - sAc[s]);
        Ms[tid][i] = to_bf16(v);
      }
    }
    {   // X slab: 32 x 64
      int r = tid >> 3, cg = (tid & 7) * 8;
      const __bf16* xp = Xb + ((size_t)(b * Lseq + c * CH + s0 + r)) * Ii + h * Pp + cg;
#pragma unroll
      for (int j = 0; j < 8; ++j) Ns[r][cg + j] = xp[j];
    }
    __syncthreads();
    v16bf af[2];
    af[0] = frag_a_16x32(&Ms[wave * 32][0], 40, lane);
    af[1] = frag_a_16x32(&Ms[wave * 32 + 16][0], 40, lane);
#pragma unroll
    for (int j = 0; j < 4; ++j) {
      v16bf bf = frag_b_kn(&Ns[0][j * 16], 72, lane);
      acc[0][j] = wmma_bf16(af[0], bf, acc[0][j]);
      acc[1][j] = wmma_bf16(af[1], bf, acc[1][j]);
    }
    __syncthreads();
  }

  // ---- phase 2: diag(exp(Acum)) C @ prevᵀ ----
  for (int n0 = 0; n0 < Nn; n0 += 32) {
    {   // scaled C rows
      const float el = __expf(sAc[tid]);
      const float* cr = Cf + ((size_t)(b * Lseq + c * CH + tid)) * Nn + n0;
#pragma unroll
      for (int i = 0; i < 32; ++i) Ms[tid][i] = to_bf16(cr[i] * el);
    }
    {   // prevᵀ slab: Ns[kn][p] = prev[p][n0+kn]
      int p = tid & 63, ng = (tid >> 6) * 8;
      const __bf16* pp =
          prevs + ((((size_t)((b * NC + c) * HEADS + h)) * Pp) + p) * Nn + n0 + ng;
#pragma unroll
      for (int j = 0; j < 8; ++j) Ns[ng + j][p] = pp[j];
    }
    __syncthreads();
    v16bf af[2];
    af[0] = frag_a_16x32(&Ms[wave * 32][0], 40, lane);
    af[1] = frag_a_16x32(&Ms[wave * 32 + 16][0], 40, lane);
#pragma unroll
    for (int j = 0; j < 4; ++j) {
      v16bf bf = frag_b_kn(&Ns[0][j * 16], 72, lane);
      acc[0][j] = wmma_bf16(af[0], bf, acc[0][j]);
      acc[1][j] = wmma_bf16(af[1], bf, acc[1][j]);
    }
    __syncthreads();
  }

  // ---- epilogue: y = acc + x*D[h], in place over x (always in bounds) ----
  const float Dh = Dvec[h];
  const int half = lane >> 4, nn = lane & 15;
  float* yb = xy + ((size_t)(b * Lseq + c * CH + wave * 32 + 8 * half)) * Ii +
              (h * Pp + nn);
#pragma unroll
  for (int i = 0; i < 2; ++i)
#pragma unroll
    for (int v = 0; v < 8; ++v) {
      float* yr = yb + (size_t)(i * 16 + v) * Ii;
#pragma unroll
      for (int j = 0; j < 4; ++j) yr[j * 16] = acc[i][j][v] + yr[j * 16] * Dh;
    }
}

// ---------------------------------------------------------------------------
// RMS norm over I + SiLU(z) gating * norm_w; result written over z region of zx
// (so the out-proj GEMM can read it with lda = ZXW).
// ---------------------------------------------------------------------------
__global__ __launch_bounds__(256) void rms_gate_kernel(
    const float* __restrict__ y, float* __restrict__ zx,
    const float* __restrict__ norm_w) {
  const int bl = blockIdx.x, tid = threadIdx.x;
  __shared__ float red[256];
  const float* yr = y + (size_t)bl * Ii;
  float ss = 0.0f;
  for (int i = tid; i < Ii; i += 256) { float v = yr[i]; ss += v * v; }
  red[tid] = ss;
  __syncthreads();
  for (int s = 128; s > 0; s >>= 1) {
    if (tid < s) red[tid] += red[tid + s];
    __syncthreads();
  }
  const float inv = rsqrtf(red[0] / (float)Ii + 1e-5f);
  float* zr = zx + (size_t)bl * ZXW;
  for (int i = tid; i < Ii; i += 256) {
    float z = zr[i];
    float g = z / (1.0f + __expf(-z));              // SiLU(z)
    zr[i] = yr[i] * inv * g * norm_w[i];            // overwrite z in place
  }
}

// ---------------------------------------------------------------------------
extern "C" void kernel_launch(void* const* d_in, const int* in_sizes, int n_in,
                              void* d_out, int out_size, void* d_ws, size_t ws_size,
                              hipStream_t stream) {
  (void)in_sizes; (void)n_in; (void)out_size; (void)ws_size;
  const float* hs   = (const float*)d_in[0];   // (2,4096,2048)
  const float* Wi   = (const float*)d_in[1];   // (8512,2048)
  const float* cw   = (const float*)d_in[2];   // (4352,1,4)
  const float* cbv  = (const float*)d_in[3];   // (4352,)
  const float* dtb  = (const float*)d_in[4];   // (64,)
  const float* Alog = (const float*)d_in[5];   // (64,)
  const float* Dv   = (const float*)d_in[6];   // (64,)
  const float* nw   = (const float*)d_in[7];   // (4096,)
  const float* Wo   = (const float*)d_in[8];   // (2048,4096)
  float* out = (float*)d_out;

  char* wsp = (char*)d_ws;
  auto carve = [&](size_t bytes) -> char* {
    char* p = wsp;
    wsp += (bytes + 255) & ~(size_t)255;
    return p;
  };
  float*  zx     = (float*) carve((size_t)BL * ZXW * 4);              // 279 MB
  float*  xy     = (float*) carve((size_t)BL * Ii * 4);               // 134 MB
  __bf16* Xb     = (__bf16*)carve((size_t)BL * Ii * 2);               //  67 MB
  float*  Bf     = (float*) carve((size_t)BL * Nn * 4);               //   4 MB
  float*  Cf     = (float*) carve((size_t)BL * Nn * 4);               //   4 MB
  float*  dtv    = (float*) carve((size_t)BL * HEADS * 4);            //   2 MB
  float*  dAv    = (float*) carve((size_t)BL * HEADS * 4);            //   2 MB
  float*  Acum   = (float*) carve((size_t)Bb * HEADS * NC * CH * 4);  //   2 MB
  float*  G      = (float*) carve((size_t)Bb * NC * CH * CH * 4);     //   8 MB
  float*  states = (float*) carve((size_t)Bb * NC * HEADS * Pp * Nn * 4); // 67 MB
  __bf16* prevs  = (__bf16*)carve((size_t)Bb * NC * HEADS * Pp * Nn * 2); // 34 MB

  // 1) in_proj: zx[8192,8512] = hs @ Wi^T
  gemm_f32wmma_kernel<<<dim3((ZXW + 127) / 128, BL / 128, 1), 256, 0, stream>>>(
      hs, Wi, zx, BL, ZXW, Hh, Hh, 0, 0, 0);

  // 2) dt / dA
  dt_kernel<<<(BL * HEADS) / 256, 256, 0, stream>>>(zx, dtb, Alog, dtv, dAv);

  // 3) conv + SiLU + split
  conv_kernel<<<(int)(((size_t)BL * XBCW + 255) / 256), 256, 0, stream>>>(
      zx, cw, cbv, dtv, xy, Xb, Bf, Cf);

  // 4) per-chunk cumsum of dA
  cumsum_kernel<<<(Bb * HEADS * NC) / 256, 256, 0, stream>>>(dAv, Acum);

  // 5) G = C @ B^T per (b,c): batched GEMM, 32 batches of 256x256, K=128
  gemm_f32wmma_kernel<<<dim3(2, 2, Bb * NC), 256, 0, stream>>>(
      Cf, Bf, G, CH, CH, Nn, Nn,
      (long long)CH * Nn, (long long)CH * Nn, (long long)CH * CH);

  // 6) per-chunk states
  chunk_states_kernel<<<dim3(HEADS, NC, Bb), 256, 0, stream>>>(Xb, Bf, Acum, states);

  // 7) inter-chunk recurrence -> prev_states (bf16)
  prev_states_kernel<<<((size_t)Bb * HEADS * Pp * Nn) / 256, 256, 0, stream>>>(
      states, Acum, prevs);

  // 8) fused Y_diag + Y_off + x*D (in place over x)
  chunk_y_kernel<<<dim3(HEADS, NC, Bb), 256, 0, stream>>>(
      G, Xb, Cf, prevs, Acum, Dv, xy);

  // 9) RMS norm + SiLU(z) gate (writes over z region of zx)
  rms_gate_kernel<<<BL, 256, 0, stream>>>(xy, zx, nw);

  // 10) out_proj: out[8192,2048] = ynorm @ Wo^T  (A rows strided by ZXW)
  gemm_f32wmma_kernel<<<dim3(Hh / 128, BL / 128, 1), 256, 0, stream>>>(
      zx, Wo, out, BL, Hh, Ii, ZXW, 0, 0, 0);
}
